// GIN1Policy_56461640073421
// MI455X (gfx1250) — matrix operand order repaired and checked by reading, not compiled, not run on hardware
//
#include <hip/hip_runtime.h>
#include <stdint.h>
#include <math.h>

// ---------------------------------------------------------------------------
// GIN bipartite policy net on gfx1250 (MI455X).
// All 64x64 contractions run on v_wmma_f32_16x16x32_f16 (f16 in, f32 accum).
// Per-node projections hoisted out of the edge loop; edge kernel fuses
// gather + LayerNorm + ReLU + WMMA + atomic segment-sum scatter.
// ---------------------------------------------------------------------------

typedef _Float16 half_t;
typedef __attribute__((ext_vector_type(16))) _Float16 v16h;
typedef __attribute__((ext_vector_type(8)))  float    v8f;

#define NC_N  32768
#define NV_N  65536
#define NO_N  96
#define ECV_N 1048576
#define EOV_N 196608
#define G_N   32

static_assert(NC_N % 16 == 0 && NV_N % 16 == 0 && NO_N % 16 == 0, "");
static_assert(ECV_N % 16 == 0 && EOV_N % 16 == 0, "");

__device__ __forceinline__ v8f wmma_f16f32(v16h a, v16h b, v8f c) {
  // D = A(16x32 f16) * B(32x16 f16) + C(16x16 f32)
  return __builtin_amdgcn_wmma_f32_16x16x32_f16(false, a, false, b, (short)0, c,
                                                false, false);
}

// B fragment (32x16) for output cols [c0,c0+16), K rows [k0,k0+32), from
// TRANSPOSED f16 weights Wt[o*KW + k] = W[k][o]. Lane n=lane&15, h=lane>>4:
// e0..7 = W[k0+8h+i][c0+n], e8..15 = W[k0+16+8h+i][c0+n] (two 16B loads).
__device__ __forceinline__ v16h load_bfrag(const half_t* __restrict__ Wt,
                                           int KW, int c0, int k0, int n, int h) {
  typedef __attribute__((ext_vector_type(8))) _Float16 v8h;
  const half_t* p = Wt + (size_t)(c0 + n) * KW + (size_t)(k0 + 8 * h);
  v8h lo = *(const v8h*)p;
  v8h hi = *(const v8h*)(p + 16);
  return __builtin_shufflevector(lo, hi, 0,1,2,3,4,5,6,7,8,9,10,11,12,13,14,15);
}

// A fragment (16x32) from row-major f32, ld=64. Lane m=lane&15, h=lane>>4:
// e0..7  = A[row][k0+8h .. +7],  e8..15 = A[row][k0+16+8h .. +7]
__device__ __forceinline__ v16h load_afrag_f32(const float* __restrict__ A,
                                               size_t row, int k0, int h) {
  const float* p = A + row * 64 + k0 + 8 * h;
  const float* q = p + 16;
  v16h r;
#pragma unroll
  for (int i = 0; i < 8; ++i) {
    r[i]     = (half_t)p[i];
    r[8 + i] = (half_t)q[i];
  }
  return r;
}

// ---------------------------------------------------------------------------
// Weight convert+transpose: Wt[o*K + k] = (f16) W[k*64 + o]
__global__ void cvt_wt_kernel(const float* __restrict__ W,
                              half_t* __restrict__ Wt, int K) {
  int t = blockIdx.x * blockDim.x + threadIdx.x;
  if (t >= 64 * K) return;
  int o = t / K, k = t % K;
  Wt[t] = (half_t)W[(size_t)k * 64 + o];
}

// First embedding layer (din in {1,3,6}): out = relu(x @ W1 + b1), out dim 64
__global__ void emb_first_kernel(const float* __restrict__ X,
                                 const float* __restrict__ W1,
                                 const float* __restrict__ b1,
                                 float* __restrict__ out, int N, int din) {
  int t = blockIdx.x * blockDim.x + threadIdx.x;
  if (t >= N * 64) return;
  int n = t >> 6, o = t & 63;
  float s = b1[o];
  for (int i = 0; i < din; ++i) s += X[(size_t)n * din + i] * W1[i * 64 + o];
  out[t] = fmaxf(s, 0.0f);
}

// out(N x 64) = act( A0(N x 64) @ W[:64] + [A1(N x 64) @ W[64:128]] + bias )
// Wt transposed f16, KW = 64 (A1 == null) or 128 (concat GEMM).
// One wave computes a 16-row x 64-col tile.  N % 16 == 0.
__global__ void gemm64_kernel(const float* __restrict__ A0,
                              const float* __restrict__ A1,
                              const half_t* __restrict__ Wt, int KW,
                              const float* __restrict__ bias,
                              float* __restrict__ out, int N, int act) {
  int wave = (int)((blockIdx.x * blockDim.x + threadIdx.x) >> 5);
  int lane = threadIdx.x & 31;
  int r0 = wave * 16;
  if (r0 >= N) return;            // wave-uniform: EXEC all-ones for WMMA
  int n = lane & 15, h = lane >> 4;

  v16h a0 = load_afrag_f32(A0, (size_t)(r0 + n), 0, h);
  v16h a1 = load_afrag_f32(A0, (size_t)(r0 + n), 32, h);
  v16h a2 = {}, a3 = {};
  if (A1) {
    a2 = load_afrag_f32(A1, (size_t)(r0 + n), 0, h);
    a3 = load_afrag_f32(A1, (size_t)(r0 + n), 32, h);
  }
#pragma unroll
  for (int t = 0; t < 4; ++t) {
    int c0 = t * 16;
    float bv = bias ? bias[c0 + n] : 0.0f;
    v8f acc;
#pragma unroll
    for (int r = 0; r < 8; ++r) acc[r] = bv;
    acc = wmma_f16f32(a0, load_bfrag(Wt, KW, c0, 0, n, h), acc);
    acc = wmma_f16f32(a1, load_bfrag(Wt, KW, c0, 32, n, h), acc);
    if (A1) {
      acc = wmma_f16f32(a2, load_bfrag(Wt, KW, c0, 64, n, h), acc);
      acc = wmma_f16f32(a3, load_bfrag(Wt, KW, c0, 96, n, h), acc);
    }
#pragma unroll
    for (int r = 0; r < 8; ++r) {
      float v = acc[r];
      if (act) v = fmaxf(v, 0.0f);
      out[(size_t)(r0 + r + 8 * h) * 64 + c0 + n] = v;  // C: c[r]=C[r+8h][n]
    }
  }
}

// Fused edge kernel: per 16-edge tile per wave.
//   h = projR[dst] + projL[src] + ef*We ; t = relu(LN(h)) ; m = t @ Wf + bf
//   atomicAdd(agg[dst], m)
__global__ void edge_kernel(const int* __restrict__ src,
                            const int* __restrict__ dst,
                            const float* __restrict__ ef,
                            const float* __restrict__ projL,
                            const float* __restrict__ projR,
                            const float* __restrict__ We,
                            const float* __restrict__ lnw,
                            const float* __restrict__ lnb,
                            const half_t* __restrict__ Wft,
                            const float* __restrict__ bf,
                            float* __restrict__ agg, int E) {
  int wave = (int)((blockIdx.x * blockDim.x + threadIdx.x) >> 5);
  int lane = threadIdx.x & 31;
  int base = wave * 16;
  if (base >= E) return;          // wave-uniform
  int m = lane & 15, h = lane >> 4;
  int e = base + m;               // lanes m and m+16 hold the same edge
  int s = src[e], d = dst[e];
  float w = ef[e];

  // this lane's 32 columns: {16g + 8h + j : g<4, j<8} (A-frag layout)
  const float* pr = projR + (size_t)d * 64;
  const float* pl = projL + (size_t)s * 64;
  float x[32];
  float sum = 0.0f, ssq = 0.0f;
#pragma unroll
  for (int g = 0; g < 4; ++g) {
    int cg = 16 * g + 8 * h;
#pragma unroll
    for (int j = 0; j < 8; ++j) {
      int c = cg + j;
      float v = pr[c] + pl[c] + w * We[c];
      x[g * 8 + j] = v;
      sum += v;
      ssq += v * v;
    }
  }
  // row-wise LN over 64: other half of the row sits in lane ^ 16
  sum += __shfl_xor(sum, 16, 32);
  ssq += __shfl_xor(ssq, 16, 32);
  float mu = sum * (1.0f / 64.0f);
  float var = ssq * (1.0f / 64.0f) - mu * mu;
  float rstd = rsqrtf(var + 1e-5f);

  v16h a0, a1;  // k-chunks 0 and 32
#pragma unroll
  for (int g = 0; g < 4; ++g) {
    int cg = 16 * g + 8 * h;
#pragma unroll
    for (int j = 0; j < 8; ++j) {
      int c = cg + j;
      float v = (x[g * 8 + j] - mu) * rstd * lnw[c] + lnb[c];
      half_t hv = (half_t)fmaxf(v, 0.0f);
      if (g == 0)      a0[j]     = hv;
      else if (g == 1) a0[8 + j] = hv;
      else if (g == 2) a1[j]     = hv;
      else             a1[8 + j] = hv;
    }
  }
#pragma unroll
  for (int t = 0; t < 4; ++t) {
    int c0 = t * 16;
    float bv = bf[c0 + m];  // n == m
    v8f acc;
#pragma unroll
    for (int r = 0; r < 8; ++r) acc[r] = bv;
    acc = wmma_f16f32(a0, load_bfrag(Wft, 64, c0, 0, m, h), acc);
    acc = wmma_f16f32(a1, load_bfrag(Wft, 64, c0, 32, m, h), acc);
#pragma unroll
    for (int r = 0; r < 8; ++r) {
      int row = r + 8 * h;                 // edge index within tile
      int drow = __shfl(d, row, 32);       // dst of that edge
      atomicAdd(&agg[(size_t)drow * 64 + c0 + m], acc[r]);
    }
  }
}

// In-place LayerNorm over rows of 64: x = (x-mu)*rstd*w + b.  One wave/row.
__global__ void ln_rows_kernel(float* __restrict__ x, const float* __restrict__ w,
                               const float* __restrict__ b, int N) {
  int wave = (int)((blockIdx.x * blockDim.x + threadIdx.x) >> 5);
  int lane = threadIdx.x & 31;
  if (wave >= N) return;
  float* row = x + (size_t)wave * 64;
  float a = row[lane], c = row[lane + 32];
  float sum = a + c, ssq = a * a + c * c;
#pragma unroll
  for (int o = 16; o; o >>= 1) {
    sum += __shfl_xor(sum, o, 32);
    ssq += __shfl_xor(ssq, o, 32);
  }
  float mu = sum * (1.0f / 64.0f);
  float var = ssq * (1.0f / 64.0f) - mu * mu;
  float rstd = rsqrtf(var + 1e-5f);
  row[lane]      = (a - mu) * rstd * w[lane]      + b[lane];
  row[lane + 32] = (c - mu) * rstd * w[lane + 32] + b[lane + 32];
}

// Mean pooling over var_batch (atomics into 32x64 + counts)
__global__ void pool_kernel(const float* __restrict__ vf,
                            const int* __restrict__ batch,
                            float* __restrict__ pooled, float* __restrict__ cnt,
                            int N) {
  int t = blockIdx.x * blockDim.x + threadIdx.x;
  if (t >= N * 64) return;
  int n = t >> 6, c = t & 63;
  int b = batch[n];
  atomicAdd(&pooled[(size_t)b * 64 + c], vf[t]);
  if (c == 0) atomicAdd(&cnt[b], 1.0f);
}

// Head: x = pooled/cnt ; h = relu(x@W1+b1) ; out = sigmoid(h@W2+b2)
__global__ void head_kernel(const float* __restrict__ pooled,
                            const float* __restrict__ cnt,
                            const float* __restrict__ W1,
                            const float* __restrict__ b1,
                            const float* __restrict__ W2,
                            const float* __restrict__ b2,
                            float* __restrict__ out) {
  __shared__ float x[64];
  __shared__ float h1[64];
  int g = blockIdx.x, t = threadIdx.x;
  float c = fmaxf(cnt[g], 1.0f);
  x[t] = pooled[(size_t)g * 64 + t] / c;
  __syncthreads();
  float s = b1[t];
  for (int i = 0; i < 64; ++i) s += x[i] * W1[i * 64 + t];
  h1[t] = fmaxf(s, 0.0f);
  __syncthreads();
  if (t == 0) {
    float s2 = b2[0];
    for (int i = 0; i < 64; ++i) s2 += h1[i] * W2[i];
    out[g] = 1.0f / (1.0f + expf(-s2));
  }
}

// ---------------------------------------------------------------------------
extern "C" void kernel_launch(void* const* d_in, const int* in_sizes, int n_in,
                              void* d_out, int out_size, void* d_ws, size_t ws_size,
                              hipStream_t stream) {
  (void)in_sizes; (void)n_in; (void)out_size; (void)ws_size;
  const float* con_x = (const float*)d_in[0];
  const float* var_x = (const float*)d_in[1];
  const float* obj_x = (const float*)d_in[2];
  const int*   cv    = (const int*)d_in[3];   // (2, ECV): row0=con, row1=var
  const float* cv_ef = (const float*)d_in[4];
  const int*   ov    = (const int*)d_in[5];   // (2, EOV): row0=obj, row1=var
  const float* ov_ef = (const float*)d_in[6];
  const int*   vbat  = (const int*)d_in[8];
  // params pytree leaves in setup_inputs() insertion order:
  //  0..3  con {W1,b1,W2,b2}   4..7 var   8..11 obj
  //  12.. c0 {Wl,bl,We,Wr,lnf_w,lnf_b,Wf,bf,lnp_w,lnp_b,Wo1,bo1,Wo2,bo2}
  //  26.. c1   40.. c2   54.. c3
  //  68 head_W1  69 head_b1  70 head_W2  71 head_b2
  const float* const* P = (const float* const*)(d_in + 9);
  float* out = (float*)d_out;

  // ---- workspace carve (256B aligned slices) ----
  char* wp = (char*)d_ws;
  auto carve = [&](size_t bytes) -> char* {
    char* r = wp;
    wp += (bytes + 255) & ~(size_t)255;
    return r;
  };
  float* cfA   = (float*)carve((size_t)NC_N * 64 * 4);
  float* cfB   = (float*)carve((size_t)NC_N * 64 * 4);
  float* vfA   = (float*)carve((size_t)NV_N * 64 * 4);
  float* vfB   = (float*)carve((size_t)NV_N * 64 * 4);
  float* ofA   = (float*)carve((size_t)NO_N * 64 * 4);
  float* ofB   = (float*)carve((size_t)NO_N * 64 * 4);
  float* projR = (float*)carve((size_t)NV_N * 64 * 4);
  float* projL = (float*)carve((size_t)NV_N * 64 * 4);  // also emb tmp & t1
  float* aggB  = (float*)carve((size_t)NV_N * 64 * 4);
  float* pooled= (float*)carve((size_t)G_N * 64 * 4);
  float* cntB  = (float*)carve((size_t)G_N * 4);
  half_t* embW2t[3];
  for (int i = 0; i < 3; ++i) embW2t[i] = (half_t*)carve(4096 * 2);
  half_t* cw[4][5];  // per conv: Wlt, Wrt, Wft, Wo1t(64x128), Wo2t
  for (int c = 0; c < 4; ++c) {
    cw[c][0] = (half_t*)carve(4096 * 2);
    cw[c][1] = (half_t*)carve(4096 * 2);
    cw[c][2] = (half_t*)carve(4096 * 2);
    cw[c][3] = (half_t*)carve(8192 * 2);
    cw[c][4] = (half_t*)carve(4096 * 2);
  }

  auto cvt = [&](const float* W, half_t* Wt, int K) {
    int tot = 64 * K;
    cvt_wt_kernel<<<(tot + 255) / 256, 256, 0, stream>>>(W, Wt, K);
  };
  auto gemm = [&](const float* A0, const float* A1, const half_t* Wt, int KW,
                  const float* bias, float* o, int N, int act) {
    int waves = N / 16;
    int blocks = (waves * 32 + 255) / 256;
    gemm64_kernel<<<blocks, 256, 0, stream>>>(A0, A1, Wt, KW, bias, o, N, act);
  };

  // ---- convert all weights to transposed f16 ----
  cvt(P[2], embW2t[0], 64);
  cvt(P[6], embW2t[1], 64);
  cvt(P[10], embW2t[2], 64);
  for (int c = 0; c < 4; ++c) {
    int pb = 12 + 14 * c;
    cvt(P[pb + 0],  cw[c][0], 64);   // Wl
    cvt(P[pb + 3],  cw[c][1], 64);   // Wr
    cvt(P[pb + 6],  cw[c][2], 64);   // Wf
    cvt(P[pb + 10], cw[c][3], 128);  // Wo1 (128x64)
    cvt(P[pb + 12], cw[c][4], 64);   // Wo2
  }

  // ---- feature embeddings: relu(relu(x@W1+b1)@W2+b2) ----
  emb_first_kernel<<<(NC_N * 64 + 255) / 256, 256, 0, stream>>>(con_x, P[0], P[1], projL, NC_N, 1);
  gemm(projL, nullptr, embW2t[0], 64, P[3], cfA, NC_N, 1);
  emb_first_kernel<<<(NV_N * 64 + 255) / 256, 256, 0, stream>>>(var_x, P[4], P[5], projL, NV_N, 3);
  gemm(projL, nullptr, embW2t[1], 64, P[7], vfA, NV_N, 1);
  emb_first_kernel<<<(NO_N * 64 + 255) / 256, 256, 0, stream>>>(obj_x, P[8], P[9], projL, NO_N, 6);
  gemm(projL, nullptr, embW2t[2], 64, P[11], ofA, NO_N, 1);

  // ---- one bipartite GIN layer ----
  auto run_conv = [&](int ci, const float* left, int Nleft, const float* right,
                      int Nright, const int* srcI, const int* dstI,
                      const float* efp, int E, float* outRight) {
    int pb = 12 + 14 * ci;
    gemm(right, nullptr, cw[ci][0], 64, P[pb + 1], projR, Nright, 0); // right@Wl+bl
    gemm(left,  nullptr, cw[ci][1], 64, nullptr,   projL, Nleft, 0);  // left@Wr
    hipMemsetAsync(aggB, 0, (size_t)Nright * 64 * 4, stream);
    {
      int waves = E / 16;
      int blocks = (waves * 32 + 255) / 256;
      edge_kernel<<<blocks, 256, 0, stream>>>(srcI, dstI, efp, projL, projR,
                                              P[pb + 2], P[pb + 4], P[pb + 5],
                                              cw[ci][2], P[pb + 7], aggB, E);
    }
    ln_rows_kernel<<<(Nright + 7) / 8, 256, 0, stream>>>(aggB, P[pb + 8], P[pb + 9], Nright);
    // t1 = relu([ln(agg), right] @ Wo1 + bo1)  (projL is free now)
    gemm(aggB, right, cw[ci][3], 128, P[pb + 11], projL, Nright, 1);
    gemm(projL, nullptr, cw[ci][4], 64, P[pb + 13], outRight, Nright, 0);
  };

  // c0: update cf  (rev_cv: src=var row, dst=con row)
  run_conv(0, vfA, NV_N, cfA, NC_N, cv + ECV_N, cv, cv_ef, ECV_N, cfB);
  // c1: update vf  (cv: src=con row, dst=var row)
  run_conv(1, cfB, NC_N, vfA, NV_N, cv, cv + ECV_N, cv_ef, ECV_N, vfB);
  // c2: update of  (rev_ov: src=var row, dst=obj row)
  run_conv(2, vfB, NV_N, ofA, NO_N, ov + EOV_N, ov, ov_ef, EOV_N, ofB);
  // c3: update vf  (ov: src=obj row, dst=var row)
  run_conv(3, ofB, NO_N, vfB, NV_N, ov, ov + EOV_N, ov_ef, EOV_N, vfA);

  // ---- pooling + head ----
  hipMemsetAsync(pooled, 0, ((size_t)G_N * 64 + G_N) * 4, stream); // pooled+cnt contiguous (8192B slice): clears both; cnt cleared again below
  hipMemsetAsync(cntB, 0, (size_t)G_N * 4, stream);
  pool_kernel<<<(NV_N * 64 + 255) / 256, 256, 0, stream>>>(vfA, vbat, pooled, cntB, NV_N);
  head_kernel<<<G_N, 64, 0, stream>>>(pooled, cntB, P[68], P[69], P[70], P[71], out);
}
